// SirenParallel_61572651155929
// MI455X (gfx1250) — compile-verified, matrix operand order checked by reading, not run on hardware
//
#include <hip/hip_runtime.h>

typedef __attribute__((ext_vector_type(16))) _Float16 v16h;
typedef __attribute__((ext_vector_type(8)))  _Float16 h8;
typedef __attribute__((ext_vector_type(8)))  float    v8f;
typedef __attribute__((ext_vector_type(4)))  float    f4;
typedef __attribute__((ext_vector_type(2)))  float    f2;

#define N_POINTS 262144
#define N_TILES  (N_POINTS / 16)
#define O_SUB    16
#define HID      64
// sin(30*z + 30*b) == v_sin_f32( (30/2pi)*z + (30/2pi)*b )  (v_sin input in revolutions)
#define RSCALE   4.774648292756860f   /* 30 / (2*pi) */

__device__ __forceinline__ v8f wmma_f16(v16h a, v16h b, v8f c) {
  // v_wmma_f32_16x16x32_f16: D = A(16x32 f16) * B(32x16 f16) + C(16x16 f32)
  return __builtin_amdgcn_wmma_f32_16x16x32_f16(false, a, false, b, (short)0, c,
                                                false, false);
}

__device__ __forceinline__ v8f zero8() {
  v8f z;
#pragma unroll
  for (int i = 0; i < 8; ++i) z[i] = 0.0f;
  return z;
}

__device__ __forceinline__ v16h zero16h() {
  v16h z;
#pragma unroll
  for (int i = 0; i < 16; ++i) z[i] = (_Float16)0.0f;
  return z;
}

// Each wave owns one subnet `o` and a strided range of 16-point tiles.
// Layers are computed transposed: h_new^T[64,16] = W[64,64] x h^T[64,16].
// A-operand = weights (rows g = 16t + (lane&15); e[2v+j] <-> K = 32kc + 16*(v>>2) + 8*half + 2*(v&3) + j).
// The f32 D tile layout (lane l: m = l&15, g = 16t + r + 8*half) provides exactly the
// per-lane K-set the next layer's B-operand needs -> no cross-lane movement between layers.
__global__ __launch_bounds__(128, 1)
void siren_parallel_wmma(const float* __restrict__ x,
                         const float* __restrict__ W0m,
                         const float* __restrict__ b0,
                         const float* __restrict__ Wm,
                         const float* __restrict__ bm,
                         const float* __restrict__ Wl,
                         const float* __restrict__ bl,
                         float* __restrict__ out,
                         int tstride) {
  const int lane = threadIdx.x & 31;
  const int half = lane >> 4;
  const int lr   = lane & 15;
  const int wave = blockIdx.x * (blockDim.x >> 5) + (threadIdx.x >> 5);
  const int o     = wave & (O_SUB - 1);
  const int tile0 = wave >> 4;

  // ---- hidden-layer weights, register-resident in WMMA A-operand f16 layout ----
  v16h Wa[3][2][4];
#pragma unroll
  for (int L = 0; L < 3; ++L) {
    const float* wb = Wm + ((size_t)(L * O_SUB + o) * HID) * HID;
#pragma unroll
    for (int kc = 0; kc < 2; ++kc) {
#pragma unroll
      for (int t = 0; t < 4; ++t) {
        const float* row = wb + (size_t)(t * 16 + lr) * HID + kc * 32 + half * 8;
        f4 q0 = *(const f4*)(row + 0);
        f4 q1 = *(const f4*)(row + 4);
        f4 q2 = *(const f4*)(row + 16);
        f4 q3 = *(const f4*)(row + 20);
        float tf[16] = {q0.x, q0.y, q0.z, q0.w, q1.x, q1.y, q1.z, q1.w,
                        q2.x, q2.y, q2.z, q2.w, q3.x, q3.y, q3.z, q3.w};
        v16h a;
#pragma unroll
        for (int i = 0; i < 16; ++i) a[i] = (_Float16)tf[i];
        Wa[L][kc][t] = a;
      }
    }
  }

  // ---- first layer: A tiles have only K=0,1 nonzero (dim_in = 2) ----
  _Float16 w0lo[4], w0hi[4];
#pragma unroll
  for (int t = 0; t < 4; ++t) {
    f2 v = *(const f2*)(W0m + ((size_t)o * HID + t * 16 + lr) * 2);
    float a0 = (half == 0) ? v.x : 0.0f;
    float a1 = (half == 0) ? v.y : 0.0f;
    w0lo[t] = (_Float16)a0;
    w0hi[t] = (_Float16)a1;
  }

  // ---- biases, pre-scaled by 30/2pi, in D-tile layout: bs[L][t][r] = RSCALE*b[16t + 8*half + r]
  h8 bs[4][4];
#pragma unroll
  for (int L = 0; L < 4; ++L) {
    const float* bp =
        (L == 0) ? (b0 + o * HID) : (bm + (size_t)((L - 1) * O_SUB + o) * HID);
#pragma unroll
    for (int t = 0; t < 4; ++t) {
      const float* p = bp + t * 16 + half * 8;
      f4 q0 = *(const f4*)(p + 0);
      f4 q1 = *(const f4*)(p + 4);
      float tf[8] = {q0.x, q0.y, q0.z, q0.w, q1.x, q1.y, q1.z, q1.w};
      h8 hb;
#pragma unroll
      for (int i = 0; i < 8; ++i) hb[i] = (_Float16)(RSCALE * tf[i]);
      bs[L][t] = hb;
    }
  }

  // ---- last-layer weights in B-operand K ordering ----
  v16h wlv[2];
#pragma unroll
  for (int c = 0; c < 2; ++c) {
    const float* p = Wl + o * HID + c * 32 + half * 8;
    f4 q0 = *(const f4*)(p + 0);
    f4 q1 = *(const f4*)(p + 4);
    f4 q2 = *(const f4*)(p + 16);
    f4 q3 = *(const f4*)(p + 20);
    float tf[16] = {q0.x, q0.y, q0.z, q0.w, q1.x, q1.y, q1.z, q1.w,
                    q2.x, q2.y, q2.z, q2.w, q3.x, q3.y, q3.z, q3.w};
    v16h a;
#pragma unroll
    for (int i = 0; i < 16; ++i) a[i] = (_Float16)tf[i];
    wlv[c] = a;
  }
  const float blv = bl[o] + 0.5f;

  // ---- software-pipelined x load: all lanes load (halves duplicate), no exec mask ----
  const f2* xp = (const f2*)x;  // element index = ti*16 + lr
  f2 xv = xp[(size_t)tile0 * 16 + lr];

  for (int ti = tile0; ti < N_TILES; ti += tstride) {
    // prefetch next tile's x (clamped so the address is always in-bounds)
    int tn = ti + tstride;
    if (tn >= N_TILES) tn = ti;
    f2 xnext = xp[(size_t)tn * 16 + lr];

    // ---- layer 0: B = x^T with only rows K=0,1 nonzero ----
    float x0 = (half == 0) ? xv.x : 0.0f;
    float x1 = (half == 0) ? xv.y : 0.0f;
    v16h bx = zero16h();
    bx[0] = (_Float16)x0;
    bx[1] = (_Float16)x1;

    v8f d[4];
#pragma unroll
    for (int t = 0; t < 4; ++t) {
      v16h a = zero16h();
      a[0]   = w0lo[t];
      a[1]   = w0hi[t];
      d[t]   = wmma_f16(a, bx, zero8());
    }

    // activation + repack: D tiles -> next layer's B operand (per-lane only)
    v16h bc0, bc1;
#pragma unroll
    for (int t = 0; t < 4; ++t)
#pragma unroll
      for (int r = 0; r < 8; ++r) {
        float s = __builtin_amdgcn_sinf(fmaf(RSCALE, d[t][r], (float)bs[0][t][r]));
        _Float16 h = (_Float16)s;
        if (t < 2) bc0[(t & 1) * 8 + r] = h;
        else       bc1[(t & 1) * 8 + r] = h;
      }

    // ---- 3 hidden layers: 8 WMMAs each ----
#pragma unroll
    for (int L = 0; L < 3; ++L) {
#pragma unroll
      for (int t = 0; t < 4; ++t) {
        v8f c = wmma_f16(Wa[L][0][t], bc0, zero8());
        d[t]  = wmma_f16(Wa[L][1][t], bc1, c);
      }
#pragma unroll
      for (int t = 0; t < 4; ++t)
#pragma unroll
        for (int r = 0; r < 8; ++r) {
          float s =
              __builtin_amdgcn_sinf(fmaf(RSCALE, d[t][r], (float)bs[L + 1][t][r]));
          _Float16 h = (_Float16)s;
          if (t < 2) bc0[(t & 1) * 8 + r] = h;
          else       bc1[(t & 1) * 8 + r] = h;
        }
    }

    // ---- last layer: per-lane 32-term dot, then combine the two half-lane g-sets
    float sum = 0.0f;
#pragma unroll
    for (int i = 0; i < 16; ++i)
      sum = fmaf((float)bc0[i], (float)wlv[0][i], sum);
#pragma unroll
    for (int i = 0; i < 16; ++i)
      sum = fmaf((float)bc1[i], (float)wlv[1][i], sum);
    sum += __shfl_xor(sum, 16, 32);
    if (half == 0) out[(size_t)(ti * 16 + lr) * O_SUB + o] = sum + blv;

    xv = xnext;
  }
}

extern "C" void kernel_launch(void* const* d_in, const int* in_sizes, int n_in,
                              void* d_out, int out_size, void* d_ws, size_t ws_size,
                              hipStream_t stream) {
  (void)in_sizes; (void)n_in; (void)out_size; (void)d_ws; (void)ws_size;
  const float* x   = (const float*)d_in[0];
  const float* W0m = (const float*)d_in[1];
  const float* b0  = (const float*)d_in[2];
  const float* Wm  = (const float*)d_in[3];
  const float* bm  = (const float*)d_in[4];
  const float* Wl  = (const float*)d_in[5];
  const float* bl  = (const float*)d_in[6];
  float* out = (float*)d_out;

  const int threads = 128;               // 4 waves/block
  const int blocks  = 2048;              // 8192 waves = 16 subnets x 512 tile streams
  const int tstride = (blocks * (threads / 32)) / O_SUB;  // 512 -> 32 tiles/wave

  hipLaunchKernelGGL(siren_parallel_wmma, dim3(blocks), dim3(threads), 0, stream,
                     x, W0m, b0, Wm, bm, Wl, bl, out, tstride);
}
